// NetVLADLayer_4157528342878
// MI455X (gfx1250) — matrix operand order repaired and checked by reading, not compiled, not run on hardware
//
#include <hip/hip_runtime.h>
#include <stdint.h>

// ---------------- problem constants ----------------
#define B_   32
#define C_   512
#define N_   1600            // H*W
#define NP_  1664            // N padded to multiple of 128 for GEMM2 n-loop
#define K_   64

typedef __attribute__((ext_vector_type(16))) __bf16 v16bf;
typedef __attribute__((ext_vector_type(8)))  float  v8f;
typedef __attribute__((ext_vector_type(4)))  unsigned int v4u;
typedef __attribute__((ext_vector_type(8)))  int v8i;
typedef __attribute__((ext_vector_type(4)))  int v4i;

union Frag { v16bf v; uint4 q[2]; };

// A operand: 16(M) x 32(K) bf16, row-major in LDS with row stride lda (bf16 elems).
// ISA layout: lane = m + 16h; elements 0-7 = K[8h..8h+7], elements 8-15 = K[16+8h..23+8h].
__device__ __forceinline__ Frag load_a_frag(const __bf16* base, int lda, int m0, int k0, int lane) {
  Frag f;
  int h = lane >> 4, m = lane & 15;
  const __bf16* p = base + (m0 + m) * lda + k0 + 8 * h;
  f.q[0] = *(const uint4*)p;
  f.q[1] = *(const uint4*)(p + 16);
  return f;
}

// B operand: 32(K) x 16(N) bf16, stored COLUMN-contiguous (i.e. BT[n][k], row stride ldb).
// ISA layout: lane = n + 16h; elements 0-7 = K[8h..8h+7], elements 8-15 = K[16+8h..23+8h].
__device__ __forceinline__ Frag load_b_frag(const __bf16* base, int ldb, int n0, int k0, int lane) {
  Frag f;
  int h = lane >> 4, n = lane & 15;
  const __bf16* p = base + (n0 + n) * ldb + k0 + 8 * h;
  f.q[0] = *(const uint4*)p;
  f.q[1] = *(const uint4*)(p + 16);
  return f;
}

// ---- TDM: DMA a rows x cols bf16 tile (in-bounds) from global into LDS ----
// LDS image gets 16B padding appended per 256B row (pad_interval=64 DW, pad_amount=4 DW),
// producing the 136-element row stride the fragment loaders expect.
__device__ __forceinline__ void tdm_load_tile(const void* gptr, uint32_t lds_addr,
                                              uint32_t rows, uint32_t cols,
                                              uint64_t row_stride_elems) {
  uint64_t ga = (uint64_t)(uintptr_t)gptr;
  v4u g0;
  g0[0] = 1u;                                               // count=1, user mode, no gather
  g0[1] = lds_addr;                                         // LDS byte address
  g0[2] = (uint32_t)ga;                                     // global_addr[31:0]
  g0[3] = ((uint32_t)(ga >> 32) & 0x1FFFFFFu) | (2u << 30); // global_addr[56:32] | type=2
  v8i g1;
  g1[0] = (int)((1u << 16)        // data_size = 2 bytes
              | (1u << 20)        // pad_enable
              | (5u << 22)        // pad_interval: 64 DWORDs (256B)
              | (3u << 25));      // pad_amount: 4 DWORDs (16B)
  g1[1] = (int)((cols & 0xFFFFu) << 16);                        // tensor_dim0[15:0]
  g1[2] = (int)(((cols >> 16) & 0xFFFFu) | ((rows & 0xFFFFu) << 16)); // dim0 hi | dim1 lo
  g1[3] = (int)(((rows >> 16) & 0xFFFFu) | ((cols & 0xFFFFu) << 16)); // dim1 hi | tile_dim0
  g1[4] = (int)(rows & 0xFFFFu);                                 // tile_dim1 | tile_dim2=0
  g1[5] = (int)(uint32_t)row_stride_elems;                       // dim0_stride[31:0]
  g1[6] = (int)(uint32_t)((row_stride_elems >> 32) & 0xFFFFu);   // dim0_stride[47:32]
  g1[7] = 0;
  v4i z4 = {0, 0, 0, 0};
  v8i z8 = {0, 0, 0, 0, 0, 0, 0, 0};
  __builtin_amdgcn_tensor_load_to_lds(g0, g1, z4, z4, z8, 0);
}

// ---------------- 0: zero accumulators + pad regions ----------------
__global__ void k_zero(float* __restrict__ asum, float* __restrict__ gssq,
                       __bf16* __restrict__ a_mat, __bf16* __restrict__ xn_cn) {
  int i = blockIdx.x * blockDim.x + threadIdx.x;
  int stride = gridDim.x * blockDim.x;
  if (i < B_ * K_) asum[i] = 0.f;
  if (i < B_)      gssq[i] = 0.f;
  const int padw = NP_ - N_;                       // 64
  for (int j = i; j < B_ * K_ * padw; j += stride) {
    int row = j / padw, col = j % padw;
    a_mat[(size_t)row * NP_ + N_ + col] = (__bf16)0.f;
  }
  for (int j = i; j < B_ * C_ * padw; j += stride) {
    int row = j / padw, col = j % padw;
    xn_cn[(size_t)row * NP_ + N_ + col] = (__bf16)0.f;
  }
}

// ---------------- 1: assign_w fp32 -> bf16 ----------------
__global__ void k_prepw(const float* __restrict__ w, __bf16* __restrict__ wbf) {
  int i = blockIdx.x * blockDim.x + threadIdx.x;
  if (i < K_ * C_) wbf[i] = (__bf16)w[i];
}

// ---------------- 2: per-pixel inverse L2 norm over channels ----------------
__global__ void k_norms(const float* __restrict__ x, float* __restrict__ invn) {
  int b = blockIdx.x;
  const float* xb = x + (size_t)b * C_ * N_;
  for (int n = threadIdx.x; n < N_; n += blockDim.x) {
    float ss = 0.f;
    for (int c = 0; c < C_; ++c) { float v = xb[(size_t)c * N_ + n]; ss += v * v; }
    invn[b * N_ + n] = 1.f / fmaxf(sqrtf(ss), 1e-12f);
  }
}

// ---------------- 3: normalized x in BOTH bf16 layouts ([C][NP] and [N][C]) ----------------
__launch_bounds__(256)
__global__ void k_dual(const float* __restrict__ x, const float* __restrict__ invn,
                       __bf16* __restrict__ xn_cn, __bf16* __restrict__ xn_nc) {
  __shared__ float tile[64][65];
  int b = blockIdx.z, c0 = blockIdx.y * 64, n0 = blockIdx.x * 64;
  int tid = threadIdx.x;
  int r = tid >> 2, q = (tid & 3) * 16;            // row + 16-col slice of 64x64 tile
  const float* src = x + ((size_t)(b * C_ + c0 + r)) * N_ + n0 + q;
  __bf16* dcn = xn_cn + ((size_t)(b * C_ + c0 + r)) * NP_ + n0 + q;
  #pragma unroll
  for (int j = 0; j < 16; ++j) {
    float v = src[j] * invn[b * N_ + n0 + q + j];
    dcn[j] = (__bf16)v;
    tile[r][q + j] = v;
  }
  __syncthreads();
  __bf16* dnc = xn_nc + ((size_t)(b * N_ + n0 + r)) * C_ + c0 + q;
  #pragma unroll
  for (int j = 0; j < 16; ++j) dnc[j] = (__bf16)tile[q + j][r];
}

// ---------------- 4: GEMM1 (logits = W * xhat) + softmax over K + a, sum_n a ----------------
__launch_bounds__(256)
__global__ void k_gemm1_softmax(const __bf16* __restrict__ xn_nc,
                                const __bf16* __restrict__ wbf,
                                __bf16* __restrict__ a_out,
                                float* __restrict__ asum) {
  __shared__ __align__(16) __bf16 Wt[64 * 136];    // [K=64][128c], TDM-padded rows
  __shared__ __align__(16) __bf16 Xt[64 * 136];    // [64 pixels][128c], TDM-padded rows
  __shared__ float lt[64 * 68];                    // logits transposed [n][k]
  int b = blockIdx.y, n0 = blockIdx.x * 64;
  int tid = threadIdx.x, lane = tid & 31, wv = tid >> 5;
  int t0 = wv, t1 = wv + 8;                        // 16 output tiles (4 m x 4 n), 2 per wave
  int mt0 = t0 >> 2, nt0 = t0 & 3, mt1 = t1 >> 2, nt1 = t1 & 3;
  v8f acc0 = {}; v8f acc1 = {};
  const __bf16* xrow = xn_nc + ((size_t)b * N_ + n0) * C_;
  for (int cc = 0; cc < C_; cc += 128) {
    if (wv == 0) {                                 // one wave drives the Tensor Data Mover
      tdm_load_tile(&wbf[cc],  (uint32_t)(uintptr_t)Wt, 64, 128, C_);
      tdm_load_tile(&xrow[cc], (uint32_t)(uintptr_t)Xt, 64, 128, C_);
      __builtin_amdgcn_s_wait_tensorcnt(0);
    }
    __syncthreads();
    #pragma unroll
    for (int kk = 0; kk < 128; kk += 32) {
      Frag a0 = load_a_frag(Wt, 136, mt0 * 16, kk, lane);
      Frag b0 = load_b_frag(Xt, 136, nt0 * 16, kk, lane);
      acc0 = __builtin_amdgcn_wmma_f32_16x16x32_bf16(false, a0.v, false, b0.v,
                                                     (short)0, acc0, false, false);
      Frag a1 = load_a_frag(Wt, 136, mt1 * 16, kk, lane);
      Frag b1 = load_b_frag(Xt, 136, nt1 * 16, kk, lane);
      acc1 = __builtin_amdgcn_wmma_f32_16x16x32_bf16(false, a1.v, false, b1.v,
                                                     (short)0, acc1, false, false);
    }
    __syncthreads();
  }
  {  // D layout: lane = ncol + 16h, element r -> row M = r + 8h
    int h = lane >> 4, nc = lane & 15;
    #pragma unroll
    for (int r = 0; r < 8; ++r) {
      lt[(nt0 * 16 + nc) * 68 + mt0 * 16 + r + 8 * h] = acc0[r];
      lt[(nt1 * 16 + nc) * 68 + mt1 * 16 + r + 8 * h] = acc1[r];
    }
  }
  __syncthreads();
  if (tid < 64) {                                  // softmax over K per pixel
    float* row = &lt[tid * 68];
    float m = -3.4e38f;
    for (int k = 0; k < K_; ++k) m = fmaxf(m, row[k]);
    float s = 0.f;
    for (int k = 0; k < K_; ++k) { float e = __expf(row[k] - m); row[k] = e; s += e; }
    float inv = 1.f / s;
    int n = n0 + tid;
    for (int k = 0; k < K_; ++k) {
      float v = row[k] * inv; row[k] = v;
      a_out[((size_t)(b * K_ + k)) * NP_ + n] = (__bf16)v;
    }
  }
  __syncthreads();
  if (tid < 64) {                                  // per-block partial sum_n a[k,n]
    float s = 0.f;
    for (int n = 0; n < 64; ++n) s += lt[n * 68 + tid];
    atomicAdd(&asum[b * K_ + tid], s);
  }
}

// ---------------- 5: GEMM2 (vlad = a * xhat^T) - asum*centroid ----------------
__launch_bounds__(256)
__global__ void k_gemm2(const __bf16* __restrict__ a_mat,
                        const __bf16* __restrict__ xn_cn,
                        const float* __restrict__ asum,
                        const float* __restrict__ centroids,
                        float* __restrict__ vlad) {
  __shared__ __align__(16) __bf16 At[64 * 136];    // a[K=64][128n], TDM-padded rows
  __shared__ __align__(16) __bf16 Xt[64 * 136];    // xhat[64c][128n], TDM-padded rows
  int b = blockIdx.y, c0 = blockIdx.x * 64;
  int tid = threadIdx.x, lane = tid & 31, wv = tid >> 5;
  int t0 = wv, t1 = wv + 8;
  int mt0 = t0 >> 2, nt0 = t0 & 3, mt1 = t1 >> 2, nt1 = t1 & 3;
  v8f acc0 = {}; v8f acc1 = {};
  const __bf16* arow = a_mat + (size_t)b * K_ * NP_;
  const __bf16* xrow = xn_cn + ((size_t)b * C_ + c0) * NP_;
  for (int nc = 0; nc < NP_; nc += 128) {          // 13 chunks over padded N
    if (wv == 0) {
      tdm_load_tile(&arow[nc], (uint32_t)(uintptr_t)At, 64, 128, NP_);
      tdm_load_tile(&xrow[nc], (uint32_t)(uintptr_t)Xt, 64, 128, NP_);
      __builtin_amdgcn_s_wait_tensorcnt(0);
    }
    __syncthreads();
    #pragma unroll
    for (int kk = 0; kk < 128; kk += 32) {
      Frag a0 = load_a_frag(At, 136, mt0 * 16, kk, lane);
      Frag b0 = load_b_frag(Xt, 136, nt0 * 16, kk, lane);
      acc0 = __builtin_amdgcn_wmma_f32_16x16x32_bf16(false, a0.v, false, b0.v,
                                                     (short)0, acc0, false, false);
      Frag a1 = load_a_frag(At, 136, mt1 * 16, kk, lane);
      Frag b1 = load_b_frag(Xt, 136, nt1 * 16, kk, lane);
      acc1 = __builtin_amdgcn_wmma_f32_16x16x32_bf16(false, a1.v, false, b1.v,
                                                     (short)0, acc1, false, false);
    }
    __syncthreads();
  }
  int h = lane >> 4, ncol = lane & 15;
  #pragma unroll
  for (int r = 0; r < 8; ++r) {
    int k = mt0 * 16 + r + 8 * h, c = c0 + nt0 * 16 + ncol;
    vlad[((size_t)(b * K_) + k) * C_ + c] = acc0[r] - asum[b * K_ + k] * centroids[k * C_ + c];
    k = mt1 * 16 + r + 8 * h; c = c0 + nt1 * 16 + ncol;
    vlad[((size_t)(b * K_) + k) * C_ + c] = acc1[r] - asum[b * K_ + k] * centroids[k * C_ + c];
  }
}

// ---------------- 6: intra-normalization per (b,k) + gather global ssq ----------------
__launch_bounds__(256)
__global__ void k_intra(const float* __restrict__ vlad, float* __restrict__ out,
                        float* __restrict__ gssq) {
  __shared__ float red[8];
  int bk = blockIdx.x;                              // b*K_ + k
  const float* v = vlad + (size_t)bk * C_;
  int tid = threadIdx.x;
  float v0 = v[tid], v1 = v[tid + 256];
  float ss = v0 * v0 + v1 * v1;
  #pragma unroll
  for (int o = 16; o; o >>= 1) ss += __shfl_down(ss, o, 32);
  if ((tid & 31) == 0) red[tid >> 5] = ss;
  __syncthreads();
  if (tid == 0) { float t = 0.f; for (int i = 0; i < 8; ++i) t += red[i]; red[0] = t; }
  __syncthreads();
  float tot = red[0];
  float inv = 1.f / fmaxf(sqrtf(tot), 1e-12f);
  out[(size_t)bk * C_ + tid]       = v0 * inv;
  out[(size_t)bk * C_ + tid + 256] = v1 * inv;
  if (tid == 0) atomicAdd(&gssq[bk >> 6], tot * inv * inv);
}

// ---------------- 7: global L2 scale ----------------
__global__ void k_final(float* __restrict__ out, const float* __restrict__ gssq) {
  size_t i = (size_t)blockIdx.x * blockDim.x + threadIdx.x;
  if (i < (size_t)B_ * K_ * C_) {
    int b = (int)(i >> 15);                         // / (K_*C_) = 32768
    out[i] *= 1.f / fmaxf(sqrtf(gssq[b]), 1e-12f);
  }
}

// ---------------- launcher ----------------
extern "C" void kernel_launch(void* const* d_in, const int* in_sizes, int n_in,
                              void* d_out, int out_size, void* d_ws, size_t ws_size,
                              hipStream_t stream) {
  (void)in_sizes; (void)n_in; (void)out_size; (void)ws_size;
  const float* x         = (const float*)d_in[0];   // [B,C,H,W]
  const float* centroids = (const float*)d_in[1];   // [K,C]
  const float* assign_w  = (const float*)d_in[2];   // [K,C]
  float* out = (float*)d_out;                       // [B, K*C]

  char* w = (char*)d_ws;
  size_t off = 0;
  auto nxt = [&](size_t bytes) -> void* {
    void* p = w + off;
    off = (off + bytes + 255) & ~(size_t)255;
    return p;
  };
  __bf16* xn_nc = (__bf16*)nxt((size_t)B_ * N_ * C_ * 2);   // 52.4 MB
  __bf16* xn_cn = (__bf16*)nxt((size_t)B_ * C_ * NP_ * 2);  // 54.5 MB
  __bf16* a_mat = (__bf16*)nxt((size_t)B_ * K_ * NP_ * 2);  // 6.8 MB
  __bf16* wbf   = (__bf16*)nxt((size_t)K_ * C_ * 2);
  float*  vlad  = (float*)nxt((size_t)B_ * K_ * C_ * 4);    // 4.2 MB
  float*  invn  = (float*)nxt((size_t)B_ * N_ * 4);
  float*  asum  = (float*)nxt((size_t)B_ * K_ * 4);
  float*  gssq  = (float*)nxt((size_t)B_ * 4);

  k_zero <<<256, 256, 0, stream>>>(asum, gssq, a_mat, xn_cn);
  k_prepw<<<(K_ * C_ + 255) / 256, 256, 0, stream>>>(assign_w, wbf);
  k_norms<<<B_, 256, 0, stream>>>(x, invn);
  k_dual <<<dim3(N_ / 64, C_ / 64, B_), 256, 0, stream>>>(x, invn, xn_cn, xn_nc);
  k_gemm1_softmax<<<dim3(N_ / 64, B_), 256, 0, stream>>>(xn_nc, wbf, a_mat, asum);
  k_gemm2<<<dim3(C_ / 64, B_), 256, 0, stream>>>(a_mat, xn_cn, asum, centroids, vlad);
  k_intra<<<B_ * K_, 256, 0, stream>>>(vlad, out, gssq);
  k_final<<<(B_ * K_ * C_ + 255) / 256, 256, 0, stream>>>(out, gssq);
}
